// MambaEEGModel_1932735283981
// MI455X (gfx1250) — matrix operand order repaired and checked by reading, not compiled, not run on hardware
//
#include <hip/hip_runtime.h>
#include <hip/hip_bf16.h>

// ---------------------------------------------------------------------------
// MambaEEG forward for MI455X (gfx1250, wave32, WMMA).
// Big GEMMs: bf16 WMMA 16x16x32, f32 accum; W tile staged via
// global_load_async_to_lds_b128 (ASYNCcnt), A tile register-pipelined.
// Attention: flash-style, f32 WMMA 16x16x4.
// SSM scan: one wave per (batch, head), 32x32 state in registers.
// ---------------------------------------------------------------------------

typedef __attribute__((ext_vector_type(16))) __bf16 v16bf;
typedef __attribute__((ext_vector_type(8)))  float  v8f;
typedef __attribute__((ext_vector_type(2)))  float  v2f;

struct __attribute__((aligned(16))) B128 { unsigned long long lo, hi; };

#define DEV_INLINE __device__ __forceinline__

DEV_INLINE float geluf(float x) {
    return 0.5f * x * (1.0f + erff(x * 0.7071067811865475f));
}
DEV_INLINE float siluf(float x) {
    return x / (1.0f + __expf(-x));
}

// ---------------------------------------------------------------------------
// fp32 -> bf16 weight conversion with N/K zero padding.
// ---------------------------------------------------------------------------
__global__ void k_cvt(const float* __restrict__ src, __bf16* __restrict__ dst,
                      int N, int K, int Npad, int Kpad) {
    int idx = blockIdx.x * 256 + threadIdx.x;
    int total = Npad * Kpad;
    if (idx >= total) return;
    int r = idx / Kpad, k = idx % Kpad;
    float v = (r < N && k < K) ? src[(size_t)r * K + k] : 0.0f;
    dst[idx] = (__bf16)v;
}

// ---------------------------------------------------------------------------
// Generic GEMM (used only for the strided spatial einsum).
// Branch-free guarded loads: clamp address, unconditional load, select-zero.
// Tile 128(M) x 64(N); 8 waves, 2x2 WMMA tiles per wave.
// ---------------------------------------------------------------------------
__global__ __launch_bounds__(256)
void k_gemm_bf16(const float* __restrict__ A, long long aBS, int aRS, int aCS,
                 const __bf16* __restrict__ W, int Kpad,
                 const float* __restrict__ bias,
                 const float* __restrict__ R, long long rBS,
                 float* __restrict__ C, long long cBS,
                 int M, int N, int K, int act) {
    __shared__ __attribute__((aligned(16))) __bf16 As[128 * 40];
    __shared__ __attribute__((aligned(16))) __bf16 Ws[64 * 40];

    int z = blockIdx.z;
    const float* Ab = A + (size_t)z * aBS;
    const float* Rb = R ? (R + (size_t)z * rBS) : nullptr;
    float* Cb = C + (size_t)z * cBS;

    int mBase = blockIdx.y * 128;
    int nBase = blockIdx.x * 64;
    int tid = threadIdx.x;
    int wave = tid >> 5, lane = tid & 31;
    int hlf = lane >> 4, l16 = lane & 15;
    int wm = wave >> 1, wn = wave & 1;

    v8f acc[2][2] = {};

    int arow = tid >> 3;
    int ak0 = (tid & 7) * 4;
    int wrow = tid >> 2;
    int wseg = (tid & 3) * 8;

    for (int kb = 0; kb < Kpad; kb += 32) {
        #pragma unroll
        for (int rr = 0; rr < 4; ++rr) {
            int row = arow + rr * 32;
            int gm = mBase + row;
            int gmc = gm < M ? gm : M - 1;
            union { __bf16 h[4]; unsigned long long u; } pk;
            #pragma unroll
            for (int j = 0; j < 4; ++j) {
                int k = kb + ak0 + j;
                int kc = k < K ? k : K - 1;
                float v = Ab[(size_t)gmc * aRS + (size_t)kc * aCS];
                pk.h[j] = (__bf16)((gm < M && k < K) ? v : 0.0f);
            }
            *(unsigned long long*)&As[row * 40 + ak0] = pk.u;
        }
        {
            const __bf16* src = W + (size_t)(nBase + wrow) * Kpad + kb + wseg;
            B128 w = *(const B128*)src;
            *(B128*)&Ws[wrow * 40 + wseg] = w;
            if (kb + 32 < Kpad) __builtin_prefetch(src + 32, 0, 1);
        }
        __syncthreads();

        union AF { v16bf v; B128 q[2]; };
        AF fa[2], fb[2];
        #pragma unroll
        for (int i = 0; i < 2; ++i) {
            int row = wm * 32 + i * 16 + l16;
            int ab = hlf * 8;
            fa[i].q[0] = *(const B128*)&As[row * 40 + ab];
            fa[i].q[1] = *(const B128*)&As[row * 40 + ab + 16];
        }
        #pragma unroll
        for (int j = 0; j < 2; ++j) {
            int n = wn * 32 + j * 16 + l16;
            int kb2 = hlf * 8;
            fb[j].q[0] = *(const B128*)&Ws[n * 40 + kb2];
            fb[j].q[1] = *(const B128*)&Ws[n * 40 + kb2 + 16];
        }
        #pragma unroll
        for (int i = 0; i < 2; ++i)
            #pragma unroll
            for (int j = 0; j < 2; ++j)
                acc[i][j] = __builtin_amdgcn_wmma_f32_16x16x32_bf16(
                    false, fa[i].v, false, fb[j].v, (short)0, acc[i][j],
                    false, false);
        __syncthreads();
    }

    #pragma unroll
    for (int i = 0; i < 2; ++i)
        #pragma unroll
        for (int j = 0; j < 2; ++j)
            #pragma unroll
            for (int v = 0; v < 8; ++v) {
                int row = mBase + wm * 32 + i * 16 + v + 8 * hlf;
                int col = nBase + wn * 32 + j * 16 + l16;
                if (row < M && col < N) {
                    float x = acc[i][j][v];
                    if (bias) x += bias[col];
                    if (Rb) x += Rb[(size_t)row * N + col];
                    if (act == 1) x = geluf(x);
                    Cb[(size_t)row * N + col] = x;
                }
            }
}

// ---------------------------------------------------------------------------
// Fast GEMM: A row-major fp32 (lda = K = Kpad), M % 128 == 0, Kpad % 32 == 0.
// Block tile 128x128, wave tile 32x64 -> 8 WMMAs per wave per K-step.
// W tile: async copy global->LDS (double buffered). A tile: reg-pipelined.
// ---------------------------------------------------------------------------
__global__ __launch_bounds__(256)
void k_gemm_fast(const float* __restrict__ A, int lda,
                 const __bf16* __restrict__ W, int Kpad,
                 const float* __restrict__ bias,
                 const float* __restrict__ R,
                 float* __restrict__ C, int M, int N, int act) {
    __shared__ __attribute__((aligned(16))) __bf16 As[128 * 40];
    __shared__ __attribute__((aligned(16))) __bf16 Ws[2][128 * 40];

    const int mBase = blockIdx.y * 128;
    const int nBase = blockIdx.x * 128;
    const int tid = threadIdx.x;
    const int wave = tid >> 5, lane = tid & 31;
    const int hlf = lane >> 4, l16 = lane & 15;
    const int wm = wave >> 1, wn = wave & 1;

    const int arow = tid >> 3;      // 0..31
    const int ak0 = (tid & 7) * 4;  // 0..28
    const int nk = Kpad / 32;

    float4 av[4];

    auto asyncW = [&](int kb, int p) {
        #pragma unroll
        for (int u = 0; u < 2; ++u) {
            int unit = tid + u * 256;          // 512 units of 16B
            int row = unit >> 2;               // 0..127
            int seg = (unit & 3) * 8;          // bf16 elems
            const __bf16* g = W + (size_t)(nBase + row) * Kpad + kb + seg;
            unsigned dst = (unsigned)(size_t)&Ws[p][row * 40 + seg];
            asm volatile("global_load_async_to_lds_b128 %0, %1, off"
                         :: "v"(dst), "v"(g) : "memory");
        }
    };
    auto loadA = [&](int kb) {
        #pragma unroll
        for (int rr = 0; rr < 4; ++rr) {
            int gm = mBase + arow + rr * 32;
            av[rr] = *(const float4*)(A + (size_t)gm * lda + kb + ak0);
        }
    };
    auto storeA = [&]() {
        #pragma unroll
        for (int rr = 0; rr < 4; ++rr) {
            union { __bf16 h[4]; unsigned long long u; } pk;
            pk.h[0] = (__bf16)av[rr].x;
            pk.h[1] = (__bf16)av[rr].y;
            pk.h[2] = (__bf16)av[rr].z;
            pk.h[3] = (__bf16)av[rr].w;
            *(unsigned long long*)&As[(arow + rr * 32) * 40 + ak0] = pk.u;
        }
    };

    asyncW(0, 0);
    loadA(0);

    v8f acc[2][4] = {};
    union AF { v16bf v; B128 q[2]; };

    for (int it = 0; it < nk; ++it) {
        int p = it & 1;
        storeA();
        asm volatile("s_wait_asynccnt 0x0" ::: "memory");
        __syncthreads();
        if (it + 1 < nk) {
            asyncW((it + 1) * 32, p ^ 1);   // overlap W fetch with WMMA
            loadA((it + 1) * 32);           // overlap A fetch with WMMA
        }
        AF fa[2], fb[4];
        #pragma unroll
        for (int i = 0; i < 2; ++i) {
            int row = wm * 32 + i * 16 + l16;
            int ab = hlf * 8;
            fa[i].q[0] = *(const B128*)&As[row * 40 + ab];
            fa[i].q[1] = *(const B128*)&As[row * 40 + ab + 16];
        }
        const __bf16* Wb = &Ws[p][0];
        #pragma unroll
        for (int j = 0; j < 4; ++j) {
            int n = wn * 64 + j * 16 + l16;
            int kb2 = hlf * 8;
            fb[j].q[0] = *(const B128*)&Wb[n * 40 + kb2];
            fb[j].q[1] = *(const B128*)&Wb[n * 40 + kb2 + 16];
        }
        #pragma unroll
        for (int i = 0; i < 2; ++i)
            #pragma unroll
            for (int j = 0; j < 4; ++j)
                acc[i][j] = __builtin_amdgcn_wmma_f32_16x16x32_bf16(
                    false, fa[i].v, false, fb[j].v, (short)0, acc[i][j],
                    false, false);
        __syncthreads();
    }

    #pragma unroll
    for (int i = 0; i < 2; ++i)
        #pragma unroll
        for (int j = 0; j < 4; ++j)
            #pragma unroll
            for (int v = 0; v < 8; ++v) {
                int row = mBase + wm * 32 + i * 16 + v + 8 * hlf;
                int col = nBase + wn * 64 + j * 16 + l16;
                if (col < N) {
                    float x = acc[i][j][v];
                    if (bias) x += bias[col];
                    if (R) x += R[(size_t)row * N + col];
                    if (act == 1) x = geluf(x);
                    C[(size_t)row * N + col] = x;
                }
            }
}

// ---------------------------------------------------------------------------
// Flash attention, dh=32, 8 heads. One block (8 waves) per (batch, head).
// ---------------------------------------------------------------------------
#define AT_LP 208
#define AT_KS (AT_LP * 33)
#define AT_SMEM_FLOATS (2 * AT_KS + 8 * 16 * 20)

__global__ __launch_bounds__(256)
void k_attn(const float* __restrict__ qkv, float* __restrict__ out, int L_) {
    extern __shared__ float sm[];
    float* Ks = sm;
    float* Vs = sm + AT_KS;
    float* Ps = Vs + AT_KS;

    int b = blockIdx.x >> 3;
    int hd = blockIdx.x & 7;
    int tid = threadIdx.x;

    for (int idx = tid; idx < AT_LP * 32; idx += 256) {
        int r = idx >> 5, c = idx & 31;
        float kv = 0.0f, vv = 0.0f;
        if (r < L_) {
            const float* rowp = qkv + (size_t)(b * L_ + r) * 768;
            kv = rowp[256 + hd * 32 + c];
            vv = rowp[512 + hd * 32 + c];
        }
        Ks[r * 33 + c] = kv;
        Vs[r * 33 + c] = vv;
    }
    __syncthreads();

    int wave = tid >> 5, lane = tid & 31;
    int hlf = lane >> 4, l16 = lane & 15;
    float* Pw = Ps + wave * 16 * 20;
    const float scale = 0.17677669529663687f; // 1/sqrt(32)
    int nkt = (L_ + 15) >> 4;

    for (int pass = 0; pass < 2; ++pass) {
        int qt = wave + pass * 8;
        bool valid = qt * 16 < L_;
        int qtc = valid ? qt : (nkt - 1);

        v2f qa[8];
        {
            int m = qtc * 16 + l16;
            if (m >= L_) m = L_ - 1;
            const float* qrow = qkv + (size_t)(b * L_ + m) * 768 + hd * 32;
            #pragma unroll
            for (int kc = 0; kc < 8; ++kc) {
                int k0 = kc * 4 + hlf * 2;
                qa[kc].x = qrow[k0];
                qa[kc].y = qrow[k0 + 1];
            }
        }

        v8f o0 = {}, o1 = {};
        float mrun[8], lrun[8];
        #pragma unroll
        for (int v = 0; v < 8; ++v) { mrun[v] = -3e38f; lrun[v] = 0.0f; }

        for (int kt = 0; kt < nkt; ++kt) {
            v8f S = {};
            #pragma unroll
            for (int kc = 0; kc < 8; ++kc) {
                v2f bf;
                int kk = kc * 4 + hlf * 2;
                const float* kr = Ks + (kt * 16 + l16) * 33 + kk;
                bf.x = kr[0];
                bf.y = kr[1];
                S = __builtin_amdgcn_wmma_f32_16x16x4_f32(
                    false, qa[kc], false, bf, (short)0, S, false, false);
            }
            bool colOOB = (kt * 16 + l16) >= L_;
            #pragma unroll
            for (int v = 0; v < 8; ++v)
                S[v] = colOOB ? -3e38f : S[v] * scale;

            #pragma unroll
            for (int v = 0; v < 8; ++v) {
                float mv = S[v];
                for (int off = 1; off < 16; off <<= 1)
                    mv = fmaxf(mv, __shfl_xor(mv, off, 32));
                float mnew = fmaxf(mrun[v], mv);
                float corr = __expf(mrun[v] - mnew);
                float p = __expf(S[v] - mnew);
                float ps = p;
                for (int off = 1; off < 16; off <<= 1)
                    ps += __shfl_xor(ps, off, 32);
                lrun[v] = lrun[v] * corr + ps;
                o0[v] *= corr;
                o1[v] *= corr;
                mrun[v] = mnew;
                Pw[(v + 8 * hlf) * 20 + l16] = p;
            }
            asm volatile("s_wait_dscnt 0" ::: "memory");

            #pragma unroll
            for (int kc = 0; kc < 4; ++kc) {
                int kk = kc * 4 + hlf * 2;
                v2f pa;
                pa.x = Pw[l16 * 20 + kk];
                pa.y = Pw[l16 * 20 + kk + 1];
                const float* vr0 = Vs + (kt * 16 + kk) * 33;
                const float* vr1 = Vs + (kt * 16 + kk + 1) * 33;
                v2f vb0, vb1;
                vb0.x = vr0[l16];       vb0.y = vr1[l16];
                vb1.x = vr0[16 + l16];  vb1.y = vr1[16 + l16];
                o0 = __builtin_amdgcn_wmma_f32_16x16x4_f32(
                    false, pa, false, vb0, (short)0, o0, false, false);
                o1 = __builtin_amdgcn_wmma_f32_16x16x4_f32(
                    false, pa, false, vb1, (short)0, o1, false, false);
            }
            asm volatile("" ::: "memory");
        }

        if (valid) {
            #pragma unroll
            for (int v = 0; v < 8; ++v) {
                int row = qt * 16 + v + 8 * hlf;
                if (row < L_) {
                    float inv = 1.0f / lrun[v];
                    float* orow = out + (size_t)(b * L_ + row) * 256 + hd * 32;
                    orow[l16]      = o0[v] * inv;
                    orow[16 + l16] = o1[v] * inv;
                }
            }
        }
    }
}

// ---------------------------------------------------------------------------
// LayerNorm (width 256), one wave per row.
// ---------------------------------------------------------------------------
__global__ __launch_bounds__(256)
void k_layernorm(const float* __restrict__ X, float* __restrict__ Y,
                 const float* __restrict__ g, const float* __restrict__ bt,
                 int nrows) {
    int wave = threadIdx.x >> 5, lane = threadIdx.x & 31;
    int row = blockIdx.x * 8 + wave;
    if (row >= nrows) return;
    const float* x = X + (size_t)row * 256;
    float v[8], s = 0.0f;
    #pragma unroll
    for (int i = 0; i < 8; ++i) { v[i] = x[lane + i * 32]; s += v[i]; }
    for (int off = 1; off < 32; off <<= 1) s += __shfl_xor(s, off, 32);
    float mean = s * (1.0f / 256.0f);
    float q = 0.0f;
    #pragma unroll
    for (int i = 0; i < 8; ++i) { float d = v[i] - mean; q += d * d; }
    for (int off = 1; off < 32; off <<= 1) q += __shfl_xor(q, off, 32);
    float inv = rsqrtf(q * (1.0f / 256.0f) + 1e-5f);
    float* y = Y + (size_t)row * 256;
    #pragma unroll
    for (int i = 0; i < 8; ++i) {
        int c = lane + i * 32;
        y[c] = (v[i] - mean) * inv * g[c] + bt[c];
    }
}

// ---------------------------------------------------------------------------
// y = (y * silu(z)) * rsqrt(mean(sq)+eps) * w ; width 512, z stride 1104.
// ---------------------------------------------------------------------------
__global__ __launch_bounds__(256)
void k_gate_rms(float* __restrict__ Y, const float* __restrict__ Z,
                const float* __restrict__ w, int nrows) {
    int wave = threadIdx.x >> 5, lane = threadIdx.x & 31;
    int row = blockIdx.x * 8 + wave;
    if (row >= nrows) return;
    float* y = Y + (size_t)row * 512;
    const float* z = Z + (size_t)row * 1104;
    float gv[16], s = 0.0f;
    #pragma unroll
    for (int i = 0; i < 16; ++i) {
        int c = lane + i * 32;
        float t = y[c] * siluf(z[c]);
        gv[i] = t;
        s += t * t;
    }
    for (int off = 1; off < 32; off <<= 1) s += __shfl_xor(s, off, 32);
    float inv = rsqrtf(s * (1.0f / 512.0f) + 1e-5f);
    #pragma unroll
    for (int i = 0; i < 16; ++i) {
        int c = lane + i * 32;
        y[c] = gv[i] * inv * w[c];
    }
}

// ---------------------------------------------------------------------------
// dt = softplus(zx[:,1088+h] + dt_bias[h]); dA = exp(dt * (-exp(A_log[h])))
// ---------------------------------------------------------------------------
__global__ void k_dtprep(const float* __restrict__ ZX,
                         const float* __restrict__ dtb,
                         const float* __restrict__ alog,
                         float* __restrict__ dt, float* __restrict__ dA,
                         int nrows) {
    int idx = blockIdx.x * 256 + threadIdx.x;
    if (idx >= nrows * 16) return;
    int row = idx >> 4, h = idx & 15;
    float x = ZX[(size_t)row * 1104 + 1088 + h] + dtb[h];
    float sp = (x > 20.0f) ? x : log1pf(__expf(x));
    float Ah = -__expf(alog[h]);
    dt[idx] = sp;
    dA[idx] = __expf(sp * Ah);
}

// ---------------------------------------------------------------------------
// Depthwise causal conv (width 4) + bias + silu.
// ---------------------------------------------------------------------------
__global__ void k_dwconv(const float* __restrict__ ZX,
                         const float* __restrict__ w,
                         const float* __restrict__ bias,
                         float* __restrict__ out, int B_, int L_) {
    int idx = blockIdx.x * 256 + threadIdx.x;
    int total = B_ * L_ * 576;
    if (idx >= total) return;
    int ch = idx % 576;
    int t = (idx / 576) % L_;
    int b = idx / (576 * L_);
    float acc = bias[ch];
    #pragma unroll
    for (int j = 0; j < 4; ++j) {
        int tt = t - 3 + j;
        if (tt >= 0)
            acc += w[ch * 4 + j] * ZX[(size_t)(b * L_ + tt) * 1104 + 512 + ch];
    }
    out[idx] = siluf(acc);
}

// ---------------------------------------------------------------------------
// SSM scan: one wave per (b, head); lane = p; 32-float state per lane.
// ---------------------------------------------------------------------------
__global__ __launch_bounds__(256)
void k_scan(const float* __restrict__ xbc, const float* __restrict__ dt,
            const float* __restrict__ dA, const float* __restrict__ Dp,
            float* __restrict__ Y, int B_, int L_) {
    int wave = threadIdx.x >> 5, lane = threadIdx.x & 31;
    int pair = blockIdx.x * 8 + wave;
    if (pair >= B_ * 16) return;
    int b = pair >> 4, h = pair & 15;
    float Dh = Dp[h];
    float hs[32];
    #pragma unroll
    for (int n = 0; n < 32; ++n) hs[n] = 0.0f;
    for (int t = 0; t < L_; ++t) {
        size_t base = (size_t)(b * L_ + t);
        const float* row = xbc + base * 576;
        float x = row[h * 32 + lane];
        float dtv = dt[base * 16 + h];
        float dAv = dA[base * 16 + h];
        float s = dtv * x;
        const float4* B4 = (const float4*)(row + 512);
        const float4* C4 = (const float4*)(row + 544);
        float y = 0.0f;
        #pragma unroll
        for (int q = 0; q < 8; ++q) {
            float4 Bq = B4[q], Cq = C4[q];
            int n = q * 4;
            hs[n + 0] = hs[n + 0] * dAv + s * Bq.x; y += hs[n + 0] * Cq.x;
            hs[n + 1] = hs[n + 1] * dAv + s * Bq.y; y += hs[n + 1] * Cq.y;
            hs[n + 2] = hs[n + 2] * dAv + s * Bq.z; y += hs[n + 2] * Cq.z;
            hs[n + 3] = hs[n + 3] * dAv + s * Bq.w; y += hs[n + 3] * Cq.w;
        }
        Y[base * 512 + h * 32 + lane] = y + Dh * x;
    }
}

// ---------------------------------------------------------------------------
// Mean-pool over time: (B,L,256) -> (B,256)
// ---------------------------------------------------------------------------
__global__ __launch_bounds__(256)
void k_pool(const float* __restrict__ H, float* __restrict__ P, int L_) {
    int b = blockIdx.x, d = threadIdx.x;
    float s = 0.0f;
    for (int t = 0; t < L_; ++t)
        s += H[(size_t)(b * L_ + t) * 256 + d];
    P[b * 256 + d] = s / (float)L_;
}

// ---------------------------------------------------------------------------
// Head: out[b] = gelu(pooled @ w1^T + b1) @ w2^T + b2
// ---------------------------------------------------------------------------
__global__ __launch_bounds__(128)
void k_head(const float* __restrict__ P, const float* __restrict__ w1,
            const float* __restrict__ b1, const float* __restrict__ w2,
            const float* __restrict__ b2, float* __restrict__ out) {
    __shared__ float sh[128];
    int b = blockIdx.x, j = threadIdx.x;
    const float* p = P + b * 256;
    const float* w = w1 + j * 256;
    float s = b1[j];
    for (int d = 0; d < 256; ++d) s += p[d] * w[d];
    sh[j] = geluf(s) * w2[j];
    __syncthreads();
    for (int st = 64; st > 0; st >>= 1) {
        if (j < st) sh[j] += sh[j + st];
        __syncthreads();
    }
    if (j == 0) out[b] = sh[0] + b2[0];
}

// ---------------------------------------------------------------------------
// Host orchestration
// ---------------------------------------------------------------------------
enum {
    IN_X = 0,
    SP_CONV_W = 1, SP_CONV_B = 2,
    SP_WQKV = 3, SP_BQKV = 4, SP_WO = 5, SP_BO = 6,
    SP_LN1_G = 7, SP_LN1_B = 8,
    SP_W1 = 9, SP_B1 = 10, SP_W2 = 11, SP_B2 = 12,
    SP_LN2_G = 13, SP_LN2_B = 14,
    NORM_IN_G = 15, NORM_IN_B = 16,
    LAYER0 = 17, // +10/layer: ln_g,ln_b,in_proj,conv_w,conv_b,dt_bias,A_log,D,rms_w,out_proj
    POST_WQKV = 77, POST_BQKV = 78, POST_WO = 79, POST_BO = 80,
    POST_LN_G = 81, POST_LN_B = 82,
    HEAD_W1 = 83, HEAD_B1 = 84, HEAD_W2 = 85, HEAD_B2 = 86
};

static constexpr int BN = 128, LSEQ = 200, BL = BN * LSEQ;
static constexpr size_t OFF_H  = 0;
static constexpr size_t OFF_A  = OFF_H  + (size_t)BL * 256  * 4;
static constexpr size_t OFF_B  = OFF_A  + (size_t)BL * 1104 * 4;
static constexpr size_t OFF_C  = OFF_B  + (size_t)BL * 576  * 4;
static constexpr size_t OFF_DT = OFF_C  + (size_t)BL * 512  * 4;
static constexpr size_t OFF_DA = OFF_DT + (size_t)BL * 16   * 4;
static constexpr size_t OFF_PL = OFF_DA + (size_t)BL * 16   * 4;
static constexpr size_t OFF_W  = OFF_PL + (size_t)128 * 256 * 4;

extern "C" void kernel_launch(void* const* d_in, const int* in_sizes, int n_in,
                              void* d_out, int out_size, void* d_ws,
                              size_t ws_size, hipStream_t stream) {
    (void)in_sizes; (void)n_in; (void)out_size; (void)ws_size;
    auto in = [&](int i) { return (const float*)d_in[i]; };

    char* ws = (char*)d_ws;
    float* bh = (float*)(ws + OFF_H);
    float* ba = (float*)(ws + OFF_A);
    float* bb = (float*)(ws + OFF_B);
    float* bc = (float*)(ws + OFF_C);
    float* bdt = (float*)(ws + OFF_DT);
    float* bda = (float*)(ws + OFF_DA);
    float* bpool = (float*)(ws + OFF_PL);
    __bf16* wbase = (__bf16*)(ws + OFF_W);

    size_t wcur = 0;
    auto cvt = [&](int idx, int N, int K, int Npad, int Kpad) -> __bf16* {
        __bf16* dst = wbase + wcur;
        wcur += (size_t)Npad * Kpad;
        int total = Npad * Kpad;
        k_cvt<<<(total + 255) / 256, 256, 0, stream>>>(in(idx), dst, N, K,
                                                       Npad, Kpad);
        return dst;
    };

    __bf16* W_conv   = cvt(SP_CONV_W, 256, 129, 256, 160);
    __bf16* W_qkv_sp = cvt(SP_WQKV, 768, 256, 768, 256);
    __bf16* W_wo_sp  = cvt(SP_WO, 256, 256, 256, 256);
    __bf16* W_w1     = cvt(SP_W1, 512, 256, 512, 256);
    __bf16* W_w2     = cvt(SP_W2, 256, 512, 256, 512);
    __bf16* W_in[6];
    __bf16* W_out[6];
    for (int i = 0; i < 6; ++i) {
        W_in[i]  = cvt(LAYER0 + 10 * i + 2, 1104, 256, 1152, 256);
        W_out[i] = cvt(LAYER0 + 10 * i + 9, 256, 512, 256, 512);
    }
    __bf16* W_qkv_po = cvt(POST_WQKV, 768, 256, 768, 256);
    __bf16* W_wo_po  = cvt(POST_WO, 256, 256, 256, 256);

    // fast GEMM: A row-major (lda == Kpad), M % 128 == 0
    auto gemmf = [&](const float* A, int lda, const __bf16* W, int Kpad,
                     const float* bias, const float* R, float* Cp,
                     int M, int N, int Npad, int act) {
        dim3 g(Npad / 128, M / 128, 1);
        k_gemm_fast<<<g, 256, 0, stream>>>(A, lda, W, Kpad, bias, R, Cp,
                                           M, N, act);
    };

    const size_t ATTN_SMEM = AT_SMEM_FLOATS * sizeof(float);

    auto mha = [&](const __bf16* Wq, int bqkv_i, const __bf16* Wo_, int bo_i,
                   int lng, int lnb) {
        gemmf(bh, 256, Wq, 256, in(bqkv_i), nullptr, ba, BL, 768, 768, 0);
        k_attn<<<BN * 8, 256, ATTN_SMEM, stream>>>(ba, bb, LSEQ);
        gemmf(bb, 256, Wo_, 256, in(bo_i), bh, bc, BL, 256, 256, 0);
        k_layernorm<<<BL / 8, 256, 0, stream>>>(bc, bh, in(lng), in(lnb), BL);
    };

    // ---- 1. spatial channel-mix (strided A -> generic kernel) ----
    {
        dim3 g(256 / 64, (200 + 127) / 128, BN);
        k_gemm_bf16<<<g, 256, 0, stream>>>(
            in(IN_X), (long long)129 * 200, 1, 200, W_conv, 160,
            in(SP_CONV_B), nullptr, 0, bh, (long long)200 * 256,
            200, 256, 129, 0);
    }

    // ---- 2. spatial MHA + LN1 ----
    mha(W_qkv_sp, SP_BQKV, W_wo_sp, SP_BO, SP_LN1_G, SP_LN1_B);

    // ---- 3. FFN (gelu) + LN2 + norm_in ----
    gemmf(bh, 256, W_w1, 256, in(SP_B1), nullptr, ba, BL, 512, 512, 1);
    gemmf(ba, 512, W_w2, 512, in(SP_B2), bh, bc, BL, 256, 256, 0);
    k_layernorm<<<BL / 8, 256, 0, stream>>>(bc, bh, in(SP_LN2_G), in(SP_LN2_B), BL);
    k_layernorm<<<BL / 8, 256, 0, stream>>>(bh, bh, in(NORM_IN_G), in(NORM_IN_B), BL);

    // ---- 4. Mamba2 layers ----
    for (int i = 0; i < 6; ++i) {
        int base = LAYER0 + 10 * i;
        k_layernorm<<<BL / 8, 256, 0, stream>>>(bh, bc, in(base + 0),
                                                in(base + 1), BL);
        gemmf(bc, 256, W_in[i], 256, nullptr, nullptr, ba, BL, 1104, 1152, 0);
        k_dtprep<<<(BL * 16 + 255) / 256, 256, 0, stream>>>(
            ba, in(base + 5), in(base + 6), bdt, bda, BL);
        k_dwconv<<<((size_t)BL * 576 + 255) / 256, 256, 0, stream>>>(
            ba, in(base + 3), in(base + 4), bb, BN, LSEQ);
        k_scan<<<(BN * 16) / 8, 256, 0, stream>>>(bb, bdt, bda, in(base + 7),
                                                  bc, BN, LSEQ);
        k_gate_rms<<<BL / 8, 256, 0, stream>>>(bc, ba, in(base + 8), BL);
        gemmf(bc, 512, W_out[i], 512, nullptr, bh, bh, BL, 256, 256, 0);
    }

    // ---- 5. post MHA + post LN ----
    mha(W_qkv_po, POST_BQKV, W_wo_po, POST_BO, POST_LN_G, POST_LN_B);

    // ---- 6. pool + head ----
    k_pool<<<BN, 256, 0, stream>>>(bh, bpool, LSEQ);
    k_head<<<BN, 128, 0, stream>>>(bpool, in(HEAD_W1), in(HEAD_B1),
                                   in(HEAD_W2), in(HEAD_B2), (float*)d_out);
}
